// MultiHeadAttention_78967268704402
// MI455X (gfx1250) — matrix operand order repaired and checked by reading, not compiled, not run on hardware
//
#include <hip/hip_runtime.h>

// ---------------- problem constants ----------------
#define DIMV   512
#define NHEADS 8
#define HD     64
#define SEQ    2048
#define BATCH  4
#define QSCALE 0.125f   // HEAD_DIM ** -0.5

// ---------------- vector types ----------------
typedef _Float16 half8  __attribute__((ext_vector_type(8)));
typedef _Float16 half16 __attribute__((ext_vector_type(16)));
typedef float    float8 __attribute__((ext_vector_type(8)));
typedef float    float4v __attribute__((ext_vector_type(4)));

// ---------------- CDNA5 async global->LDS copy (inline asm) ----------------
// global_load_async_to_lds_b128: VDST = LDS byte offset (wave-relative),
// VADDR = 64-bit global address, SADDR = off (GV mode). Tracked by ASYNCcnt.
// Generic pointers to __shared__ are {aperture | lds_offset}; low 32 bits are
// the wave-relative LDS address (ISA 10.2), so truncation gives VDST directly.
static __device__ __forceinline__ void copy16_to_lds(_Float16* lds, const _Float16* g) {
  unsigned lofs = (unsigned)(size_t)lds;   // low 32 bits = LDS offset
  asm volatile("global_load_async_to_lds_b128 %0, %1, off"
               :: "v"(lofs), "v"(g) : "memory");
}

static __device__ __forceinline__ void wait_async_lds() {
  asm volatile("s_wait_asynccnt 0x0" ::: "memory");
}

static __device__ __forceinline__ half16 cat8(half8 lo, half8 hi) {
  return __builtin_shufflevector(lo, hi, 0,1,2,3,4,5,6,7,8,9,10,11,12,13,14,15);
}

static __device__ __forceinline__ float8 wmma16(half16 a, half16 b, float8 c) {
  // D = A(16x32 f16) * B(32x16 f16) + C(16x16 f32)
  return __builtin_amdgcn_wmma_f32_16x16x32_f16(false, a, false, b, (short)0, c,
                                                false, false);
}

// ---------------- f32 -> f16 convert (weights) ----------------
__global__ void k_cvt_f16(const float* __restrict__ s, _Float16* __restrict__ d, int n) {
  int i = blockIdx.x * blockDim.x + threadIdx.x;
  if (i < n) d[i] = (_Float16)s[i];
}

// ---------------- QKV projection: out = (X @ W^T + b) * scale ----------------
// X: f32 [8192, 512] row-major.  W: f16 [512(out j), 512(in d)] row-major.
// W tiles double-buffered in LDS via async copies (shared by all 8 waves).
// Output written f16, head-major: [(b*H+h), s, d].
__global__ __launch_bounds__(256) void k_proj_qkv(const float* __restrict__ X,
    const _Float16* __restrict__ Wh, const float* __restrict__ bias,
    _Float16* __restrict__ outHM, float scale)
{
  __shared__ __attribute__((aligned(16))) _Float16 Wsh[2][64][40];

  const int tid  = threadIdx.x;
  const int wave = tid >> 5, lane = tid & 31, lh = lane & 15;
  const int hsel = (lane >> 4) & 1;
  const int m0 = blockIdx.x * 128 + wave * 16;
  const int n0 = blockIdx.y * 64;

  const float* ap = X + (size_t)(m0 + lh) * DIMV;

  // staging assignment: 64 rows x 4 chunks of 8 halfs = 256 threads
  const int wrow = tid >> 2;            // 0..63
  const int wcol = (tid & 3) * 8;       // 0..24
  const _Float16* wsrc = Wh + (size_t)(n0 + wrow) * DIMV + wcol;

  float8 zf = {0,0,0,0,0,0,0,0};
  float8 acc0 = zf, acc1 = zf, acc2 = zf, acc3 = zf;

  // preissue first W tile
  copy16_to_lds(&Wsh[0][wrow][wcol], wsrc);

  int buf = 0;
  for (int k0 = 0; k0 < DIMV; k0 += 32) {
    wait_async_lds();
    __syncthreads();
    if (k0 + 32 < DIMV)
      copy16_to_lds(&Wsh[buf ^ 1][wrow][wcol], wsrc + (k0 + 32));

    // A fragment: lane<16 holds row lh, K {k0..k0+15}; lane>=16: {k0+16..k0+31}
    const int o0 = k0 + hsel * 16;
    float4v xa = *(const float4v*)(ap + o0);
    float4v xb = *(const float4v*)(ap + o0 + 4);
    float4v xc = *(const float4v*)(ap + o0 + 8);
    float4v xd = *(const float4v*)(ap + o0 + 12);
    half16 afrag;
#pragma unroll
    for (int j = 0; j < 4; ++j) {
      afrag[j]      = (_Float16)xa[j];
      afrag[4 + j]  = (_Float16)xb[j];
      afrag[8 + j]  = (_Float16)xc[j];
      afrag[12 + j] = (_Float16)xd[j];
    }
    const int ko = hsel * 16;
    half16 b0 = cat8(*(const half8*)&Wsh[buf][ 0 + lh][ko], *(const half8*)&Wsh[buf][ 0 + lh][ko + 8]);
    half16 b1 = cat8(*(const half8*)&Wsh[buf][16 + lh][ko], *(const half8*)&Wsh[buf][16 + lh][ko + 8]);
    half16 b2 = cat8(*(const half8*)&Wsh[buf][32 + lh][ko], *(const half8*)&Wsh[buf][32 + lh][ko + 8]);
    half16 b3 = cat8(*(const half8*)&Wsh[buf][48 + lh][ko], *(const half8*)&Wsh[buf][48 + lh][ko + 8]);
    acc0 = wmma16(afrag, b0, acc0);
    acc1 = wmma16(afrag, b1, acc1);
    acc2 = wmma16(afrag, b2, acc2);
    acc3 = wmma16(afrag, b3, acc3);
    buf ^= 1;
  }

#pragma unroll
  for (int nt = 0; nt < 4; ++nt) {
    const int j  = n0 + nt * 16 + lh;
    const float bj = bias[j];
    const int h = j >> 6, d = j & 63;
    float8 acc = (nt == 0) ? acc0 : (nt == 1) ? acc1 : (nt == 2) ? acc2 : acc3;
#pragma unroll
    for (int i = 0; i < 8; ++i) {
      const int r = m0 + i + hsel * 8;               // global row in [0, 8192)
      const int b = r >> 11, s = r & 2047;
      const float v = (acc[i] + bj) * scale;
      outHM[((size_t)(b * NHEADS + h) * SEQ + s) * HD + d] = (_Float16)v;
    }
  }
}

// ---------------- output projection: out = Ctx @ Wo^T + bo (f32 out) --------
__global__ __launch_bounds__(256) void k_proj_out(const _Float16* __restrict__ Ctx,
    const _Float16* __restrict__ Wh, const float* __restrict__ bias,
    float* __restrict__ out)
{
  __shared__ __attribute__((aligned(16))) _Float16 Wsh[2][64][40];

  const int tid  = threadIdx.x;
  const int wave = tid >> 5, lane = tid & 31, lh = lane & 15;
  const int hsel = (lane >> 4) & 1;
  const int m0 = blockIdx.x * 128 + wave * 16;
  const int n0 = blockIdx.y * 64;

  const _Float16* ap = Ctx + (size_t)(m0 + lh) * DIMV;

  const int wrow = tid >> 2;
  const int wcol = (tid & 3) * 8;
  const _Float16* wsrc = Wh + (size_t)(n0 + wrow) * DIMV + wcol;

  float8 zf = {0,0,0,0,0,0,0,0};
  float8 acc0 = zf, acc1 = zf, acc2 = zf, acc3 = zf;

  copy16_to_lds(&Wsh[0][wrow][wcol], wsrc);

  int buf = 0;
  for (int k0 = 0; k0 < DIMV; k0 += 32) {
    wait_async_lds();
    __syncthreads();
    if (k0 + 32 < DIMV)
      copy16_to_lds(&Wsh[buf ^ 1][wrow][wcol], wsrc + (k0 + 32));

    const int o0 = k0 + hsel * 16;
    half16 afrag = cat8(*(const half8*)(ap + o0), *(const half8*)(ap + o0 + 8));
    const int ko = hsel * 16;
    half16 b0 = cat8(*(const half8*)&Wsh[buf][ 0 + lh][ko], *(const half8*)&Wsh[buf][ 0 + lh][ko + 8]);
    half16 b1 = cat8(*(const half8*)&Wsh[buf][16 + lh][ko], *(const half8*)&Wsh[buf][16 + lh][ko + 8]);
    half16 b2 = cat8(*(const half8*)&Wsh[buf][32 + lh][ko], *(const half8*)&Wsh[buf][32 + lh][ko + 8]);
    half16 b3 = cat8(*(const half8*)&Wsh[buf][48 + lh][ko], *(const half8*)&Wsh[buf][48 + lh][ko + 8]);
    acc0 = wmma16(afrag, b0, acc0);
    acc1 = wmma16(afrag, b1, acc1);
    acc2 = wmma16(afrag, b2, acc2);
    acc3 = wmma16(afrag, b3, acc3);
    buf ^= 1;
  }

#pragma unroll
  for (int nt = 0; nt < 4; ++nt) {
    const int j  = n0 + nt * 16 + lh;
    const float bj = bias[j];
    float8 acc = (nt == 0) ? acc0 : (nt == 1) ? acc1 : (nt == 2) ? acc2 : acc3;
#pragma unroll
    for (int i = 0; i < 8; ++i) {
      const int r = m0 + i + hsel * 8;
      out[(size_t)r * DIMV + j] = acc[i] + bj;
    }
  }
}

// ---------------- fused flash attention (online softmax) ----------------
// Q/K/V head-major f16 [(b*H+h), s, 64]; Q already scaled.
// Block: 8 waves; wave w handles Q rows [q0, q0+16). Loop over K chunks of 32.
// K chunks double-buffered via async global->LDS; V transposed through VGPRs.
__global__ __launch_bounds__(256) void k_attn(const _Float16* __restrict__ Qh,
    const _Float16* __restrict__ Kh, const _Float16* __restrict__ Vh,
    _Float16* __restrict__ Ctx)
{
  __shared__ __attribute__((aligned(16))) _Float16 Ksh[2][32][72];  // K chunk row-major
  __shared__ __attribute__((aligned(16))) _Float16 VTsh[64][40];    // V chunk transposed
  __shared__ __attribute__((aligned(16))) _Float16 Psh[8][16][40];  // per-wave P tile

  const int tid  = threadIdx.x;
  const int wave = tid >> 5, lane = tid & 31, lh = lane & 15;
  const int hsel = (lane >> 4) & 1;
  const int bh = blockIdx.y;
  const int q0 = blockIdx.x * 128 + wave * 16;

  const size_t base = (size_t)bh * SEQ * HD;
  const _Float16* Qp = Qh + base;
  const _Float16* Kp = Kh + base;
  const _Float16* Vp = Vh + base;

  // Q A-fragments (held in registers for the whole loop)
  const _Float16* qrow = Qp + (size_t)(q0 + lh) * HD;
  const int aoff = hsel * 16;
  half16 qa0 = cat8(*(const half8*)(qrow + aoff),      *(const half8*)(qrow + aoff + 8));
  half16 qa1 = cat8(*(const half8*)(qrow + 32 + aoff), *(const half8*)(qrow + 32 + aoff + 8));

  float8 zf = {0,0,0,0,0,0,0,0};
  float8 O0 = zf, O1 = zf, O2 = zf, O3 = zf;
  float m_i[8], l_i[8];
#pragma unroll
  for (int i = 0; i < 8; ++i) { m_i[i] = -1e30f; l_i[i] = 0.0f; }

  const int srow = tid >> 3;          // 0..31 : staging row
  const int scol = (tid & 7) * 8;     // 0..56 : staging col
  const int koff = hsel * 16;

  // preissue K chunk 0
  copy16_to_lds(&Ksh[0][srow][scol], Kp + (size_t)srow * HD + scol);

  int buf = 0;
  for (int kb = 0; kb < SEQ; kb += 32) {
    // ---- stage V chunk (sync, transposed through VGPRs) ----
    half8 vvv = *(const half8*)(Vp + (size_t)(kb + srow) * HD + scol);
    __syncthreads();                    // prior reads of VTsh complete
#pragma unroll
    for (int j = 0; j < 8; ++j) VTsh[scol + j][srow] = vvv[j];

    wait_async_lds();                   // our share of K chunk landed
    __syncthreads();                    // all shares landed + VTsh ready

    // issue async copy of the next K chunk into the other buffer
    if (kb + 32 < SEQ)
      copy16_to_lds(&Ksh[buf ^ 1][srow][scol],
                    Kp + (size_t)(kb + 32 + srow) * HD + scol);

    // ---- S = Q @ K^T : two 16x16 tiles (cols kb.. and kb+16..) ----
    float8 S0 = zf, S1 = zf;
    {
      half16 b00 = cat8(*(const half8*)&Ksh[buf][lh][koff],           *(const half8*)&Ksh[buf][lh][koff + 8]);
      half16 b01 = cat8(*(const half8*)&Ksh[buf][lh][32 + koff],      *(const half8*)&Ksh[buf][lh][32 + koff + 8]);
      half16 b10 = cat8(*(const half8*)&Ksh[buf][16 + lh][koff],      *(const half8*)&Ksh[buf][16 + lh][koff + 8]);
      half16 b11 = cat8(*(const half8*)&Ksh[buf][16 + lh][32 + koff], *(const half8*)&Ksh[buf][16 + lh][32 + koff + 8]);
      S0 = wmma16(qa0, b00, S0); S0 = wmma16(qa1, b01, S0);
      S1 = wmma16(qa0, b10, S1); S1 = wmma16(qa1, b11, S1);
    }

    // ---- online softmax over the 32 new columns ----
    float alpha[8];
#pragma unroll
    for (int i = 0; i < 8; ++i) {
      float s0 = S0[i], s1 = S1[i];
      float mx = fmaxf(s0, s1);
#pragma unroll
      for (int off = 1; off < 16; off <<= 1) mx = fmaxf(mx, __shfl_xor(mx, off, 32));
      const float mn = fmaxf(m_i[i], mx);
      const float a  = __expf(m_i[i] - mn);
      const float p0 = __expf(s0 - mn);
      const float p1 = __expf(s1 - mn);
      float rs = p0 + p1;
#pragma unroll
      for (int off = 1; off < 16; off <<= 1) rs += __shfl_xor(rs, off, 32);
      l_i[i] = l_i[i] * a + rs;
      m_i[i] = mn;
      alpha[i] = a;
      const int prow = i + hsel * 8;
      Psh[wave][prow][lh]      = (_Float16)p0;
      Psh[wave][prow][16 + lh] = (_Float16)p1;
    }
#pragma unroll
    for (int i = 0; i < 8; ++i) {
      O0[i] *= alpha[i]; O1[i] *= alpha[i]; O2[i] *= alpha[i]; O3[i] *= alpha[i];
    }

    // ---- reload P in A-fragment layout (LDS round-trip, same wave) ----
    const _Float16* pr = &Psh[wave][lh][0];
    half16 pa = cat8(*(const half8*)(pr + aoff), *(const half8*)(pr + aoff + 8));

    // ---- O += P @ V ----
    half16 vb0 = cat8(*(const half8*)&VTsh[lh][koff],      *(const half8*)&VTsh[lh][koff + 8]);
    half16 vb1 = cat8(*(const half8*)&VTsh[16 + lh][koff], *(const half8*)&VTsh[16 + lh][koff + 8]);
    half16 vb2 = cat8(*(const half8*)&VTsh[32 + lh][koff], *(const half8*)&VTsh[32 + lh][koff + 8]);
    half16 vb3 = cat8(*(const half8*)&VTsh[48 + lh][koff], *(const half8*)&VTsh[48 + lh][koff + 8]);
    O0 = wmma16(pa, vb0, O0);
    O1 = wmma16(pa, vb1, O1);
    O2 = wmma16(pa, vb2, O2);
    O3 = wmma16(pa, vb3, O3);

    buf ^= 1;
  }

  // ---- normalize and store context (f16, [b,s, h*64+d] row-major) ----
  const int b = bh >> 3, h = bh & 7;
#pragma unroll
  for (int i = 0; i < 8; ++i) {
    const int s = q0 + i + hsel * 8;
    const float inv = 1.0f / l_i[i];
    _Float16* op = Ctx + (size_t)(b * SEQ + s) * DIMV + h * HD + lh;
    op[0]  = (_Float16)(O0[i] * inv);
    op[16] = (_Float16)(O1[i] * inv);
    op[32] = (_Float16)(O2[i] * inv);
    op[48] = (_Float16)(O3[i] * inv);
  }
}

// ---------------- host-side orchestration ----------------
extern "C" void kernel_launch(void* const* d_in, const int* in_sizes, int n_in,
                              void* d_out, int out_size, void* d_ws, size_t ws_size,
                              hipStream_t stream) {
  (void)in_sizes; (void)n_in; (void)out_size; (void)ws_size;

  const float* q  = (const float*)d_in[0];
  const float* k  = (const float*)d_in[1];
  const float* v  = (const float*)d_in[2];
  const float* Wq = (const float*)d_in[3];
  const float* bq = (const float*)d_in[4];
  const float* Wk = (const float*)d_in[5];
  const float* bk = (const float*)d_in[6];
  const float* Wv = (const float*)d_in[7];
  const float* bv = (const float*)d_in[8];
  const float* Wo = (const float*)d_in[9];
  const float* bo = (const float*)d_in[10];
  float* out = (float*)d_out;

  _Float16* ws = (_Float16*)d_ws;
  const size_t WE = (size_t)DIMV * DIMV;                 // 262144
  const size_t QE = (size_t)BATCH * NHEADS * SEQ * HD;   // 4194304
  _Float16* WqH = ws;
  _Float16* WkH = WqH + WE;
  _Float16* WvH = WkH + WE;
  _Float16* WoH = WvH + WE;
  _Float16* Qhm = WoH + WE;
  _Float16* Khm = Qhm + QE;
  _Float16* Vhm = Khm + QE;
  _Float16* Ctx = Vhm + QE;

  // weights -> f16 (no transpose needed: W rows are B-fragment columns)
  dim3 cg((unsigned)(WE / 256));
  k_cvt_f16<<<cg, 256, 0, stream>>>(Wq, WqH, (int)WE);
  k_cvt_f16<<<cg, 256, 0, stream>>>(Wk, WkH, (int)WE);
  k_cvt_f16<<<cg, 256, 0, stream>>>(Wv, WvH, (int)WE);
  k_cvt_f16<<<cg, 256, 0, stream>>>(Wo, WoH, (int)WE);

  // QKV projections (scale folded into Q)
  dim3 pg(BATCH * SEQ / 128, DIMV / 64);  // (64, 8)
  k_proj_qkv<<<pg, 256, 0, stream>>>(q, WqH, bq, Qhm, QSCALE);
  k_proj_qkv<<<pg, 256, 0, stream>>>(k, WkH, bk, Khm, 1.0f);
  k_proj_qkv<<<pg, 256, 0, stream>>>(v, WvH, bv, Vhm, 1.0f);

  // fused attention
  dim3 ag(SEQ / 128, BATCH * NHEADS);     // (16, 32)
  k_attn<<<ag, 256, 0, stream>>>(Qhm, Khm, Vhm, Ctx);

  // output projection
  k_proj_out<<<pg, 256, 0, stream>>>(Ctx, WoH, bo, out);
}
